// NutNet_41248865911433
// MI455X (gfx1250) — matrix operand order repaired and checked by reading, not compiled, not run on hardware
//
#include <hip/hip_runtime.h>
#include <hip/hip_bf16.h>

// ---------------- problem constants (from reference) ----------------
#define BATCH   4096
#define KST     9
#define T_LOAD  64
#define N_SRV   1000
#define N_SVC   200
#define E_EDGES 32000
#define EMB     8
#define TRA     8

typedef __attribute__((ext_vector_type(16))) __bf16 v16bf;
typedef __attribute__((ext_vector_type(8)))  float  v8f;

__device__ __forceinline__ v8f wmma_bf16(v16bf a, v16bf b, v8f c) {
    // D = A(16x32 bf16) * B(32x16 bf16) + C(16x16 f32)
    return __builtin_amdgcn_wmma_f32_16x16x32_bf16(
        /*neg_a=*/false, a, /*neg_b=*/false, b,
        /*c_mod=*/(short)0, c, /*reuse_a=*/false, /*reuse_b=*/false);
}

__device__ __forceinline__ float sigmf(float x) { return 1.f / (1.f + __expf(-x)); }

// Build an A fragment from two contiguous 8-float runs (ISA 16-bit A layout:
// element e<8 -> K0+8g+e ; e>=8 -> K0+16+8g+(e-8)).
__device__ __forceinline__ v16bf afrag(const float* p0, const float* p1) {
    v16bf a = {};
    for (int e = 0; e < 8; ++e) {
        a[e]     = (__bf16)p0[e];
        a[8 + e] = (__bf16)p1[e];
    }
    return a;
}

// ---------------- utility: zero a float buffer ----------------
__global__ void k_zero(float* p, long n) {
    long stride = (long)gridDim.x * blockDim.x;
    for (long i = (long)blockIdx.x * blockDim.x + threadIdx.x; i < n; i += stride)
        p[i] = 0.f;
}

// ---------------- embedding gathers ----------------
__global__ void k_embed(const float* e_emb, const float* s_emb,
                        const int* srv_res, const int* svc_res,
                        float* srv_emb, float* svc_emb) {
    int i = blockIdx.x * blockDim.x + threadIdx.x;
    if (i < N_SRV) {
        for (int j = 0; j < 3; ++j) {
            int r = srv_res[i * 3 + j];
            for (int e = 0; e < EMB; ++e)
                srv_emb[i * 24 + j * EMB + e] = e_emb[(j * 7 + r) * EMB + e];
        }
    }
    if (i < N_SVC) {
        for (int j = 0; j < 3; ++j) {
            int r = svc_res[i * 3 + j];
            for (int e = 0; e < EMB; ++e)
                svc_emb[i * 24 + j * EMB + e] = s_emb[(j * 5 + r) * EMB + e];
        }
    }
}

// ---------------- weight pre-packing into WMMA-fragment order ----------------
// Fragment order: idx = ((kstep*ntiles + tn)*32 + lane)*16 + e,
// value = B[(kstep*32) + (e&7) + 8*(lane>>4) + 16*(e>>3), tn*16 + (lane&15)]  (zeros if OOB).

// temporal-conv weight (Cout,Cin,1,3): B[k,co] = w[co, k%Cin, 0, k/Cin]
__global__ void k_pack_w(const float* __restrict__ w, __bf16* __restrict__ pk,
                         int Cin, int Cout) {
    int K = 3 * Cin, steps = K >> 5, ntiles = (Cout + 15) >> 4;
    int tot = steps * ntiles * 32 * 16;
    for (int i = blockIdx.x * blockDim.x + threadIdx.x; i < tot;
         i += gridDim.x * blockDim.x) {
        int e = i & 15, l = (i >> 4) & 31, q = i >> 9;
        int tn = q % ntiles, s = q / ntiles;
        int k = (s << 5) + ((e & 8) << 1) + ((l >> 4) << 3) + (e & 7);
        int co = tn * 16 + (l & 15);
        int dt = k / Cin, ci = k - dt * Cin;
        float v = (co < Cout) ? w[((long)co * Cin + ci) * 3 + dt] : 0.f;
        pk[i] = (__bf16)v;
    }
}

// chebyshev weight (3,64,64): B[k,co] = cw[k/64][k%64][co], K=192, ntiles=4
__global__ void k_pack_cheb(const float* __restrict__ cw, __bf16* __restrict__ pk) {
    int tot = 6 * 4 * 32 * 16;
    for (int i = blockIdx.x * blockDim.x + threadIdx.x; i < tot;
         i += gridDim.x * blockDim.x) {
        int e = i & 15, l = (i >> 4) & 31, q = i >> 9;
        int tn = q & 3, s = q >> 2;
        int k = (s << 5) + ((e & 8) << 1) + ((l >> 4) << 3) + (e & 7);
        int term = k >> 6, kk = k & 63;
        int co = tn * 16 + (l & 15);
        pk[i] = (__bf16)cw[((long)term * 64 + kk) * 64 + co];
    }
}

// svc_emb (200,24) -> padded (224,32), K steps=7, ntiles=2
__global__ void k_pack_svcB(const float* __restrict__ svc_emb, __bf16* __restrict__ pk) {
    int tot = 7 * 2 * 32 * 16;
    for (int i = blockIdx.x * blockDim.x + threadIdx.x; i < tot;
         i += gridDim.x * blockDim.x) {
        int e = i & 15, l = (i >> 4) & 31, q = i >> 9;
        int tn = q & 1, s = q >> 1;
        int k = (s << 5) + ((e & 8) << 1) + ((l >> 4) << 3) + (e & 7);
        int n = tn * 16 + (l & 15);
        float v = (k < 200 && n < 24) ? svc_emb[k * 24 + n] : 0.f;
        pk[i] = (__bf16)v;
    }
}

// ---------------- sequential LSTM scan over BATCH (lane-8 only matters) ----
// reference scans over axis 0 (=BATCH); [:, -1, :] keeps lane 8 => one chain
// of 4096 cell steps on tra[b, 8, :]. One wave: lane r owns gate row r.
__global__ void k_lstm(const float* __restrict__ tra,
                       const float* __restrict__ w_ih, const float* __restrict__ w_hh,
                       const float* __restrict__ b_ih, const float* __restrict__ b_hh,
                       float* __restrict__ tra_f) {
    int lane = threadIdx.x;           // blockDim == 32, wave32
    float wih[4], whh[8];
    for (int k = 0; k < 4; ++k) wih[k] = w_ih[lane * 4 + k];
    for (int k = 0; k < 8; ++k) whh[k] = w_hh[lane * 8 + k];
    float bsum = b_ih[lane] + b_hh[lane];
    float h = 0.f, c = 0.f;           // valid in lanes 0..7
    for (int b = 0; b < BATCH; ++b) {
        const float* x = tra + ((long)b * KST + 8) * 4;
        float g = bsum + wih[0] * x[0] + wih[1] * x[1] + wih[2] * x[2] + wih[3] * x[3];
        for (int k = 0; k < 8; ++k) g += whh[k] * __shfl(h, k, 32);
        float gi = __shfl(g, lane & 7, 32);
        float gf = __shfl(g, (8  + lane) & 31, 32);
        float gg = __shfl(g, (16 + lane) & 31, 32);
        float go = __shfl(g, (24 + lane) & 31, 32);
        if (lane < 8) {
            c = sigmf(gf) * c + sigmf(gi) * tanhf(gg);
            h = sigmf(go) * tanhf(c);
            tra_f[(long)b * TRA + lane] = h;
        }
    }
}

// ---------------- e_load MLP (27->16->8, linear-linear then relu), fused
// concat: writes channels [0,8) of the 32-channel st1 input x1 ----------
__global__ void k_eload(const float* __restrict__ load, const float* __restrict__ srv_emb,
                        const float* __restrict__ w0, const float* __restrict__ b0,
                        const float* __restrict__ w1, const float* __restrict__ b1,
                        float* __restrict__ x1) {
    long m = (long)blockIdx.x * blockDim.x + threadIdx.x;   // (u*9+t)*1000+n
    const long Mtot = (long)T_LOAD * KST * N_SRV;
    if (m >= Mtot) return;
    int n = (int)(m % N_SRV);
    float in[27];
    for (int j = 0; j < 3; ++j)  in[j]     = load[m * 3 + j];
    for (int j = 0; j < 24; ++j) in[3 + j] = srv_emb[n * 24 + j];
    float h1[16];
    for (int r = 0; r < 16; ++r) {
        float a = b0[r];
        for (int k = 0; k < 27; ++k) a += w0[r * 27 + k] * in[k];
        h1[r] = a;
    }
    float* out = x1 + m * 32;
    for (int r = 0; r < 8; ++r) {
        float a = b1[r];
        for (int k = 0; k < 16; ++k) a += w1[r * 16 + k] * h1[k];
        out[r] = a > 0.f ? a : 0.f;
    }
}

// ---------------- svc_tot GEMM: (576000 x 200) @ (200 x 24) via bf16 WMMA,
// fused concat: writes channels [8,32) of x1. Packed B staged in LDS. -----
__global__ void k_svcgemm(const float* __restrict__ svc, const __bf16* __restrict__ pkB,
                          float* __restrict__ x1, long Mtot) {
    __shared__ __bf16 Bs[7 * 2 * 32 * 16];              // 7168 bf16 = 14336 B
    int tid = threadIdx.x;
    {
        const uint4* src = (const uint4*)pkB;
        uint4* dst = (uint4*)Bs;
        for (int i = tid; i < (7 * 2 * 32 * 16 * 2) / 16; i += blockDim.x)
            dst[i] = src[i];
    }
    __syncthreads();

    int wave = tid >> 5, lane = tid & 31;
    int g = lane >> 4, l15 = lane & 15;
    long mbase = ((long)blockIdx.x * (blockDim.x >> 5) + wave) * 16;
    if (mbase >= Mtot) return;
    long mrow = mbase + l15; if (mrow >= Mtot) mrow = Mtot - 1;
    const float* arow = svc + mrow * 200;
    __builtin_prefetch(arow, 0, 0);                     // global_prefetch_b8

    v8f c0 = {}; v8f c1 = {};
    for (int s = 0; s < 7; ++s) {
        int K0 = s << 5;
        v16bf a;
        if (s < 6) {                                    // all K in range
            a = afrag(arow + K0 + (g << 3), arow + K0 + 16 + (g << 3));
        } else {                                        // tail: K 192..223, pad >=200
            v16bf t = {};
            for (int e = 0; e < 8; ++e) {
                int k0 = K0 + (g << 3) + e;
                int k1 = K0 + 16 + (g << 3) + e;
                t[e]     = (__bf16)(k0 < 200 ? arow[k0] : 0.f);
                t[8 + e] = (__bf16)(k1 < 200 ? arow[k1] : 0.f);
            }
            a = t;
        }
        const __bf16* bp = Bs + (((s * 2) * 32 + lane) << 4);
        v16bf b0 = *(const v16bf*)bp;                   // 2x ds_load_b128
        v16bf b1 = *(const v16bf*)(bp + 32 * 16);
        c0 = wmma_bf16(a, b0, c0);
        c1 = wmma_bf16(a, b1, c1);
    }
    for (int r = 0; r < 8; ++r) {
        long row = mbase + r + (g << 3);
        if (row < Mtot) {
            x1[row * 32 + 8 + l15] = c0[r];
            int col = 16 + l15;
            if (col < 24) x1[row * 32 + 8 + col] = c1[r];
        }
    }
}

// ---------------- temporal conv (GLU): out = relu(P * sigmoid(Q) + R) ------
// Each of P/Q/R is a GEMM with K = 3*Cin (96 or 192); packed bf16 weights.
__global__ void k_tconv(const float* __restrict__ x,
                        const __bf16* __restrict__ pk1, const float* __restrict__ b1,
                        const __bf16* __restrict__ pk2, const float* __restrict__ b2,
                        const __bf16* __restrict__ pk3, const float* __restrict__ b3,
                        float* __restrict__ out, int U, int Tin, int Cin, int Cout) {
    int Tout = Tin - 2;
    long Mtot = (long)U * Tout * N_SRV;
    int K = 3 * Cin;                                    // multiple of 32
    int lane = threadIdx.x & 31, wave = threadIdx.x >> 5;
    int ntiles = (Cout + 15) >> 4;
    long wid = (long)blockIdx.x * (blockDim.x >> 5) + wave;
    long mtile = wid / ntiles;
    int  ntile = (int)(wid % ntiles);
    long mbase = mtile * 16;
    if (mbase >= Mtot) return;
    int g = lane >> 4, l15 = lane & 15;

    long mrow = mbase + l15; if (mrow >= Mtot) mrow = Mtot - 1;
    long u = mrow / ((long)Tout * N_SRV);
    long rem = mrow - u * (long)Tout * N_SRV;
    int t = (int)(rem / N_SRV), n = (int)(rem % N_SRV);
    const float* xr = x + (((long)u * Tin + t) * N_SRV + n) * Cin;
    long dts = (long)N_SRV * Cin;                       // dt stride
    __builtin_prefetch(xr, 0, 0);

    int co = ntile * 16 + l15;
    bool co_ok = co < Cout;
    int cos = co_ok ? co : 0;

    v8f aP = {}, aQ = {}, aR = {};
    for (int K0 = 0; K0 < K; K0 += 32) {
        // A runs never straddle a dt block (Cin % 8 == 0, runs 8-aligned)
        int kA0 = K0 + (g << 3);
        int dt0 = kA0 / Cin, ci0 = kA0 - dt0 * Cin;
        int kA1 = K0 + 16 + (g << 3);
        int dt1 = kA1 / Cin, ci1 = kA1 - dt1 * Cin;
        v16bf a = afrag(xr + dt0 * dts + ci0, xr + dt1 * dts + ci1);

        int s = K0 >> 5;
        long off = (((long)(s * ntiles + ntile) * 32 + lane) << 4);
        v16bf f1 = *(const v16bf*)(pk1 + off);          // 2x global_load_b128 each
        v16bf f2 = *(const v16bf*)(pk2 + off);
        v16bf f3 = *(const v16bf*)(pk3 + off);
        aP = wmma_bf16(a, f1, aP);
        aQ = wmma_bf16(a, f2, aQ);
        aR = wmma_bf16(a, f3, aR);
    }
    float bb1 = co_ok ? b1[cos] : 0.f;
    float bb2 = co_ok ? b2[cos] : 0.f;
    float bb3 = co_ok ? b3[cos] : 0.f;
    for (int r = 0; r < 8; ++r) {
        long row = mbase + r + (g << 3);
        if (row < Mtot && co_ok) {
            float p = aP[r] + bb1;
            float q = sigmf(aQ[r] + bb2);
            float y = p * q + (aR[r] + bb3);
            out[row * (long)Cout + co] = y > 0.f ? y : 0.f;
        }
    }
}

// ---------------- graph normalization precompute ----------------
__global__ void k_deg(const int* __restrict__ ei, float* deg) {
    int e = blockIdx.x * blockDim.x + threadIdx.x;
    if (e < E_EDGES) atomicAdd(&deg[ei[e]], 1.f);      // src = ei[0][e]
}
__global__ void k_dinv(const float* deg, float* dinv) {
    int n = blockIdx.x * blockDim.x + threadIdx.x;
    if (n < N_SRV) {
        float d = deg[n];
        dinv[n] = d > 0.f ? rsqrtf(fmaxf(d, 1.f)) : 0.f;
    }
}
__global__ void k_norm(const int* __restrict__ ei, const float* __restrict__ dinv,
                       float* __restrict__ normw) {
    int e = blockIdx.x * blockDim.x + threadIdx.x;
    if (e < E_EDGES)
        normw[e] = -(dinv[ei[e]] * dinv[ei[E_EDGES + e]]);
}

// ---------------- Lhat scatter: out[b2,dst,:] += z[b2,src,:]*norm[e] -------
__global__ void k_lhat(const float* __restrict__ z, const int* __restrict__ ei,
                       const float* __restrict__ normw, float* __restrict__ out,
                       int B2, int H) {
    long idx = (long)blockIdx.x * blockDim.x + threadIdx.x;
    long tot = (long)B2 * E_EDGES;
    if (idx >= tot) return;
    int e  = (int)(idx % E_EDGES);
    long b2 = idx / E_EDGES;
    int s = ei[e], d = ei[E_EDGES + e];
    float nv = normw[e];
    const float* zr = z + (b2 * N_SRV + s) * (long)H;
    float* orow     = out + (b2 * N_SRV + d) * (long)H;
    for (int h = 0; h < H; h += 4) {
        float4 v = *(const float4*)(zr + h);
        atomicAdd(&orow[h + 0], v.x * nv);
        atomicAdd(&orow[h + 1], v.y * nv);
        atomicAdd(&orow[h + 2], v.z * nv);
        atomicAdd(&orow[h + 3], v.w * nv);
    }
}

// Tx2 = 2*Lhat(Tx1) - Tx0 (in place on the Lhat(Tx1) buffer)
__global__ void k_tx2(float* __restrict__ txc, const float* __restrict__ tx0, long n) {
    long stride = (long)gridDim.x * blockDim.x;
    for (long i = (long)blockIdx.x * blockDim.x + threadIdx.x; i < n; i += stride)
        txc[i] = 2.f * txc[i] - tx0[i];
}

// ---------------- Chebyshev combine: relu([Tx0|Tx1|Tx2] @ [w0;w1;w2] + b) --
// K = 192 (3 terms x 64), Cout = 64; packed bf16 weights.
__global__ void k_cheb(const float* __restrict__ tx0, const float* __restrict__ tx1,
                       const float* __restrict__ tx2, const __bf16* __restrict__ pk,
                       const float* __restrict__ cb, float* __restrict__ out, long Mtot) {
    int lane = threadIdx.x & 31, wave = threadIdx.x >> 5;
    long wid = (long)blockIdx.x * (blockDim.x >> 5) + wave;
    long mtile = wid >> 2;
    int  ntile = (int)(wid & 3);
    long mbase = mtile * 16;
    if (mbase >= Mtot) return;
    int g = lane >> 4, l15 = lane & 15;
    long mrow = mbase + l15; if (mrow >= Mtot) mrow = Mtot - 1;
    const float* rows[3] = { tx0 + mrow * 64, tx1 + mrow * 64, tx2 + mrow * 64 };
    int co = ntile * 16 + l15;

    v8f acc = {};
    for (int s = 0; s < 6; ++s) {
        int term = s >> 1, kbase = (s & 1) << 5;
        const float* p0 = rows[term] + kbase + (g << 3);
        v16bf a = afrag(p0, p0 + 16);
        long off = (((long)(s * 4 + ntile) * 32 + lane) << 4);
        v16bf b = *(const v16bf*)(pk + off);
        acc = wmma_bf16(a, b, acc);
    }
    float bias = cb[co];
    for (int r = 0; r < 8; ++r) {
        long row = mbase + r + (g << 3);
        if (row < Mtot) {
            float y = acc[r] + bias;
            out[row * 64 + co] = y > 0.f ? y : 0.f;
        }
    }
}

// ---------------- batch norm over (U,T,C) per server n, in place ----------
__global__ void k_bn(float* __restrict__ x, const float* __restrict__ gam,
                     const float* __restrict__ bet, int UT, int C) {
    int n = blockIdx.x;                       // one block per server
    int cnt = UT * C;
    __shared__ float s_sum[256], s_sq[256];
    float sum = 0.f, sq = 0.f;
    for (int i = threadIdx.x; i < cnt; i += blockDim.x) {
        int c = i % C, ut = i / C;
        float v = x[((long)ut * N_SRV + n) * C + c];
        sum += v; sq += v * v;
    }
    s_sum[threadIdx.x] = sum; s_sq[threadIdx.x] = sq;
    __syncthreads();
    for (int o = 128; o > 0; o >>= 1) {
        if ((int)threadIdx.x < o) {
            s_sum[threadIdx.x] += s_sum[threadIdx.x + o];
            s_sq[threadIdx.x]  += s_sq[threadIdx.x + o];
        }
        __syncthreads();
    }
    float mean = s_sum[0] / cnt;
    float var  = s_sq[0] / cnt - mean * mean;
    float sc = rsqrtf(var + 1e-5f) * gam[n];
    float sh = bet[n];
    for (int i = threadIdx.x; i < cnt; i += blockDim.x) {
        int c = i % C, ut = i / C;
        long off = ((long)ut * N_SRV + n) * C + c;
        x[off] = (x[off] - mean) * sc + sh;
    }
}

// ---------------- final gather + qos MLP (88->32->16->1) ----------------
__global__ void k_qos(const float* __restrict__ tra_f, const float* __restrict__ x2,
                      const float* __restrict__ svc_emb, const int* __restrict__ info,
                      const float* __restrict__ w0, const float* __restrict__ b0,
                      const float* __restrict__ w1, const float* __restrict__ b1,
                      const float* __restrict__ w2, const float* __restrict__ b2,
                      float* __restrict__ out) {
    int b = blockIdx.x * blockDim.x + threadIdx.x;
    if (b >= BATCH) return;
    int srv = info[b * 4 + 1], svcid = info[b * 4 + 2], t3 = info[b * 4 + 3];
    int u = t3 - KST;                                  // inv[b] (identity unique)
    float in[88];
    for (int j = 0; j < 8;  ++j) in[j]      = tra_f[(long)b * 8 + j];
    const float* xr = x2 + ((long)u * N_SRV + srv) * 56;
    for (int j = 0; j < 56; ++j) in[8 + j]  = xr[j];
    for (int j = 0; j < 24; ++j) in[64 + j] = svc_emb[svcid * 24 + j];
    float h1[32];
    for (int r = 0; r < 32; ++r) {
        float a = b0[r];
        for (int k = 0; k < 88; ++k) a += w0[r * 88 + k] * in[k];
        h1[r] = a > 0.f ? a : 0.f;
    }
    float h2[16];
    for (int r = 0; r < 16; ++r) {
        float a = b1[r];
        for (int k = 0; k < 32; ++k) a += w1[r * 32 + k] * h1[k];
        h2[r] = a > 0.f ? a : 0.f;
    }
    float a = b2[0];
    for (int k = 0; k < 16; ++k) a += w2[k] * h2[k];
    out[b] = a;
}

// ======================= host-side orchestration =======================
extern "C" void kernel_launch(void* const* d_in, const int* in_sizes, int n_in,
                              void* d_out, int out_size, void* d_ws, size_t ws_size,
                              hipStream_t stream) {
    (void)in_sizes; (void)n_in; (void)out_size; (void)ws_size;
    const float* tra     = (const float*)d_in[0];
    const int*   info    = (const int*)  d_in[1];
    const float* load    = (const float*)d_in[2];
    const float* svc     = (const float*)d_in[3];
    const int*   ei      = (const int*)  d_in[4];
    const int*   srv_res = (const int*)  d_in[5];
    const int*   svc_res = (const int*)  d_in[6];
    // params flattened in jax pytree order (dict keys sorted):
    int p = 7;
    const float* e_emb   = (const float*)d_in[p++];                   // (3,7,8)
    const float* el0_b   = (const float*)d_in[p++];                   // e_load[0] b,w
    const float* el0_w   = (const float*)d_in[p++];
    const float* el1_b   = (const float*)d_in[p++];
    const float* el1_w   = (const float*)d_in[p++];
    const float* l_bhh   = (const float*)d_in[p++];                   // lstm b_hh,b_ih,w_hh,w_ih
    const float* l_bih   = (const float*)d_in[p++];
    const float* l_whh   = (const float*)d_in[p++];
    const float* l_wih   = (const float*)d_in[p++];
    const float* q0_b    = (const float*)d_in[p++];                   // qos
    const float* q0_w    = (const float*)d_in[p++];
    const float* q1_b    = (const float*)d_in[p++];
    const float* q1_w    = (const float*)d_in[p++];
    const float* q2_b    = (const float*)d_in[p++];
    const float* q2_w    = (const float*)d_in[p++];
    const float* s_emb   = (const float*)d_in[p++];                   // (3,5,8)
    const float* s1_bnb  = (const float*)d_in[p++];                   // st1
    const float* s1_bng  = (const float*)d_in[p++];
    const float* s1_cb   = (const float*)d_in[p++];
    const float* s1_cw   = (const float*)d_in[p++];
    const float* s1t1_b1 = (const float*)d_in[p++];
    const float* s1t1_b2 = (const float*)d_in[p++];
    const float* s1t1_b3 = (const float*)d_in[p++];
    const float* s1t1_w1 = (const float*)d_in[p++];
    const float* s1t1_w2 = (const float*)d_in[p++];
    const float* s1t1_w3 = (const float*)d_in[p++];
    const float* s1t2_b1 = (const float*)d_in[p++];
    const float* s1t2_b2 = (const float*)d_in[p++];
    const float* s1t2_b3 = (const float*)d_in[p++];
    const float* s1t2_w1 = (const float*)d_in[p++];
    const float* s1t2_w2 = (const float*)d_in[p++];
    const float* s1t2_w3 = (const float*)d_in[p++];
    const float* s2_bnb  = (const float*)d_in[p++];                   // st2
    const float* s2_bng  = (const float*)d_in[p++];
    const float* s2_cb   = (const float*)d_in[p++];
    const float* s2_cw   = (const float*)d_in[p++];
    const float* s2t1_b1 = (const float*)d_in[p++];
    const float* s2t1_b2 = (const float*)d_in[p++];
    const float* s2t1_b3 = (const float*)d_in[p++];
    const float* s2t1_w1 = (const float*)d_in[p++];
    const float* s2t1_w2 = (const float*)d_in[p++];
    const float* s2t1_w3 = (const float*)d_in[p++];
    const float* s2t2_b1 = (const float*)d_in[p++];
    const float* s2t2_b2 = (const float*)d_in[p++];
    const float* s2t2_b3 = (const float*)d_in[p++];
    const float* s2t2_w1 = (const float*)d_in[p++];
    const float* s2t2_w2 = (const float*)d_in[p++];
    const float* s2t2_w3 = (const float*)d_in[p++];

    // -------- workspace carve-out --------
    float* base = (float*)d_ws;
    size_t o = 0;
    float* srv_emb = base + o; o += (size_t)N_SRV * 24;
    float* svc_emb = base + o; o += (size_t)N_SVC * 24;
    float* tra_f   = base + o; o += (size_t)BATCH * TRA;
    float* deg     = base + o; o += N_SRV;
    float* dinv    = base + o; o += N_SRV;
    float* normw   = base + o; o += E_EDGES;
    o = (o + 15) & ~(size_t)15;                        // 64B align for bf16 packs
    __bf16* pbase = (__bf16*)(base + o);
    size_t ob = 0;                                     // bf16 elements
    __bf16* pk_svc  = pbase + ob; ob += 7 * 2 * 32 * 16;     // 7168
    __bf16* pk_11a  = pbase + ob; ob += 3 * 4 * 32 * 16;     // st1.tc1 w1 (6144)
    __bf16* pk_11b  = pbase + ob; ob += 3 * 4 * 32 * 16;
    __bf16* pk_11c  = pbase + ob; ob += 3 * 4 * 32 * 16;
    __bf16* pk_12a  = pbase + ob; ob += 6 * 2 * 32 * 16;     // st1.tc2
    __bf16* pk_12b  = pbase + ob; ob += 6 * 2 * 32 * 16;
    __bf16* pk_12c  = pbase + ob; ob += 6 * 2 * 32 * 16;
    __bf16* pk_21a  = pbase + ob; ob += 3 * 4 * 32 * 16;     // st2.tc1
    __bf16* pk_21b  = pbase + ob; ob += 3 * 4 * 32 * 16;
    __bf16* pk_21c  = pbase + ob; ob += 3 * 4 * 32 * 16;
    __bf16* pk_22a  = pbase + ob; ob += 6 * 4 * 32 * 16;     // st2.tc2 (12288)
    __bf16* pk_22b  = pbase + ob; ob += 6 * 4 * 32 * 16;
    __bf16* pk_22c  = pbase + ob; ob += 6 * 4 * 32 * 16;
    __bf16* pk_c1   = pbase + ob; ob += 6 * 4 * 32 * 16;     // st1.cheb
    __bf16* pk_c2   = pbase + ob; ob += 6 * 4 * 32 * 16;     // st2.cheb
    o += (ob + 1) / 2;                                 // advance float offset
    o = (o + 15) & ~(size_t)15;
    float* x1      = base + o; o += (size_t)T_LOAD * KST * N_SRV * 32;   // 18.43M
    float* bufA    = base + o; o += (size_t)T_LOAD * 7 * N_SRV * 64;     // 28.67M
    float* bufB    = base + o; o += (size_t)T_LOAD * 7 * N_SRV * 64;
    float* bufC    = base + o; o += (size_t)T_LOAD * 7 * N_SRV * 64;
    float* bufD    = base + o; o += (size_t)T_LOAD * 7 * N_SRV * 64;

    const long Mx   = (long)T_LOAD * KST * N_SRV;       // 576000
    const long Mst1 = (long)T_LOAD * 7 * N_SRV;         // 448000
    const long Mbn1 = (long)T_LOAD * 5 * N_SRV;         // 320000
    const long Mst2 = (long)T_LOAD * 3 * N_SRV;         // 192000
    const long Mout = (long)T_LOAD * 1 * N_SRV;         // 64000

    // -------- prologue --------
    k_zero<<<64, 256, 0, stream>>>(deg, N_SRV);
    k_embed<<<(N_SRV + 255) / 256, 256, 0, stream>>>(e_emb, s_emb, srv_res, svc_res, srv_emb, svc_emb);
    k_lstm<<<1, 32, 0, stream>>>(tra, l_wih, l_whh, l_bih, l_bhh, tra_f);
    k_deg<<<(E_EDGES + 255) / 256, 256, 0, stream>>>(ei, deg);
    k_dinv<<<(N_SRV + 255) / 256, 256, 0, stream>>>(deg, dinv);
    k_norm<<<(E_EDGES + 255) / 256, 256, 0, stream>>>(ei, dinv, normw);

    // weight pre-packing (tiny)
    k_pack_svcB<<<8, 256, 0, stream>>>(svc_emb, pk_svc);
    k_pack_w<<<8, 256, 0, stream>>>(s1t1_w1, pk_11a, 32, 64);
    k_pack_w<<<8, 256, 0, stream>>>(s1t1_w2, pk_11b, 32, 64);
    k_pack_w<<<8, 256, 0, stream>>>(s1t1_w3, pk_11c, 32, 64);
    k_pack_w<<<8, 256, 0, stream>>>(s1t2_w1, pk_12a, 64, 32);
    k_pack_w<<<8, 256, 0, stream>>>(s1t2_w2, pk_12b, 64, 32);
    k_pack_w<<<8, 256, 0, stream>>>(s1t2_w3, pk_12c, 64, 32);
    k_pack_w<<<8, 256, 0, stream>>>(s2t1_w1, pk_21a, 32, 64);
    k_pack_w<<<8, 256, 0, stream>>>(s2t1_w2, pk_21b, 32, 64);
    k_pack_w<<<8, 256, 0, stream>>>(s2t1_w3, pk_21c, 32, 64);
    k_pack_w<<<8, 256, 0, stream>>>(s2t2_w1, pk_22a, 64, 56);
    k_pack_w<<<8, 256, 0, stream>>>(s2t2_w2, pk_22b, 64, 56);
    k_pack_w<<<8, 256, 0, stream>>>(s2t2_w3, pk_22c, 64, 56);
    k_pack_cheb<<<8, 256, 0, stream>>>(s1_cw, pk_c1);
    k_pack_cheb<<<8, 256, 0, stream>>>(s2_cw, pk_c2);

    k_eload<<<(int)((Mx + 255) / 256), 256, 0, stream>>>(load, srv_emb, el0_w, el0_b, el1_w, el1_b, x1);
    {   // svc_tot GEMM -> x1[:,8:32]
        long mtiles = (Mx + 15) / 16;
        k_svcgemm<<<(int)((mtiles + 7) / 8), 256, 0, stream>>>(svc, pk_svc, x1, Mx);
    }

    // -------- ST block 1: x1(64,9,1000,32) --------
    {   // tc1 -> bufA (64,7,1000,64)
        long waves = ((Mst1 + 15) / 16) * 4;
        k_tconv<<<(int)((waves + 7) / 8), 256, 0, stream>>>(
            x1, pk_11a, s1t1_b1, pk_11b, s1t1_b2, pk_11c, s1t1_b3,
            bufA, T_LOAD, 9, 32, 64);
    }
    long nb1 = Mst1 * 64;
    k_zero<<<4096, 256, 0, stream>>>(bufB, nb1);
    k_lhat<<<(int)(((long)448 * E_EDGES + 255) / 256), 256, 0, stream>>>(bufA, ei, normw, bufB, 448, 64);
    k_zero<<<4096, 256, 0, stream>>>(bufC, nb1);
    k_lhat<<<(int)(((long)448 * E_EDGES + 255) / 256), 256, 0, stream>>>(bufB, ei, normw, bufC, 448, 64);
    k_tx2<<<4096, 256, 0, stream>>>(bufC, bufA, nb1);
    {   // cheb -> bufD, relu
        long waves = ((Mst1 + 15) / 16) * 4;
        k_cheb<<<(int)((waves + 7) / 8), 256, 0, stream>>>(bufA, bufB, bufC, pk_c1, s1_cb, bufD, Mst1);
    }
    {   // tc2: bufD(64,7,1000,64) -> bufA (64,5,1000,32)
        long waves = ((Mbn1 + 15) / 16) * 2;
        k_tconv<<<(int)((waves + 7) / 8), 256, 0, stream>>>(
            bufD, pk_12a, s1t2_b1, pk_12b, s1t2_b2, pk_12c, s1t2_b3,
            bufA, T_LOAD, 7, 64, 32);
    }
    k_bn<<<N_SRV, 256, 0, stream>>>(bufA, s1_bng, s1_bnb, T_LOAD * 5, 32);

    // -------- ST block 2: bufA(64,5,1000,32) --------
    {   // tc1 -> bufB (64,3,1000,64)
        long waves = ((Mst2 + 15) / 16) * 4;
        k_tconv<<<(int)((waves + 7) / 8), 256, 0, stream>>>(
            bufA, pk_21a, s2t1_b1, pk_21b, s2t1_b2, pk_21c, s2t1_b3,
            bufB, T_LOAD, 5, 32, 64);
    }
    long nb2 = Mst2 * 64;
    k_zero<<<4096, 256, 0, stream>>>(bufC, nb2);
    k_lhat<<<(int)(((long)192 * E_EDGES + 255) / 256), 256, 0, stream>>>(bufB, ei, normw, bufC, 192, 64);
    k_zero<<<4096, 256, 0, stream>>>(bufD, nb2);
    k_lhat<<<(int)(((long)192 * E_EDGES + 255) / 256), 256, 0, stream>>>(bufC, ei, normw, bufD, 192, 64);
    k_tx2<<<4096, 256, 0, stream>>>(bufD, bufB, nb2);
    {   // cheb -> x1 (reused; 12.29M <= 18.43M)
        long waves = ((Mst2 + 15) / 16) * 4;
        k_cheb<<<(int)((waves + 7) / 8), 256, 0, stream>>>(bufB, bufC, bufD, pk_c2, s2_cb, x1, Mst2);
    }
    {   // tc2: x1(64,3,1000,64) -> bufB (64,1,1000,56)
        long waves = ((Mout + 15) / 16) * 4;
        k_tconv<<<(int)((waves + 7) / 8), 256, 0, stream>>>(
            x1, pk_22a, s2t2_b1, pk_22b, s2t2_b2, pk_22c, s2t2_b3,
            bufB, T_LOAD, 3, 64, 56);
    }
    k_bn<<<N_SRV, 256, 0, stream>>>(bufB, s2_bng, s2_bnb, T_LOAD * 1, 56);

    // -------- epilogue: gather + qos MLP --------
    k_qos<<<(BATCH + 255) / 256, 256, 0, stream>>>(
        tra_f, bufB, svc_emb, info, q0_w, q0_b, q1_w, q1_b, q2_w, q2_b, (float*)d_out);
}
